// CABAttention_74689481277741
// MI455X (gfx1250) — matrix operand order repaired and checked by reading, not compiled, not run on hardware
//
#include <hip/hip_runtime.h>

// ---------------------------------------------------------------------------
// CDNA5 / gfx1250 block-sparse attention (wave32, WMMA bf16, async->LDS GEMM)
// ---------------------------------------------------------------------------
typedef __attribute__((ext_vector_type(16))) __bf16 v16bf;
typedef __attribute__((ext_vector_type(8)))  __bf16 v8bf;
typedef __attribute__((ext_vector_type(8)))  float  v8f;

constexpr int B_  = 2;
constexpr int N_  = 2048;
constexpr int D_  = 1024;
constexpr int H_  = 16;
constexpr int HD_ = 64;
constexpr int BS_ = 64;
constexpr int M_  = 32;           // N_/BS_
constexpr float SCALE_ = 0.125f;  // HD^-0.5

// Async DMA: global -> LDS, 16B per lane, tracked by ASYNCcnt (gfx1250).
__device__ __forceinline__ void async_ld_b128(unsigned lds_addr,
                                              unsigned long long gaddr) {
  asm volatile("global_load_async_to_lds_b128 %0, %1, off"
               :: "v"(lds_addr), "v"(gaddr) : "memory");
}
__device__ __forceinline__ void wait_async0() {
  asm volatile("s_wait_asynccnt 0" ::: "memory");
}

// ---------------------------------------------------------------------------
// fp32 -> bf16 elementwise convert
// ---------------------------------------------------------------------------
__global__ void cvt_bf16_kernel(const float* __restrict__ in,
                                __bf16* __restrict__ out, int n) {
  int i = blockIdx.x * blockDim.x + threadIdx.x;
  if (i < n) out[i] = (__bf16)in[i];
}

// ---------------------------------------------------------------------------
// GEMM: C[M,N] (f32, +optional bias) = A[M,K] bf16 * B[N,K]^T bf16
// Workgroup (256 thr / 8 waves, 2x4 wave grid) computes a 64x128 macro-tile.
// Per K-step (32): A 64x32 + B 128x32 staged in LDS via async global->LDS DMA,
// double-buffered; next step's DMA overlaps current step's WMMAs.
// Each wave: 32x32 tile = 2x2 WMMA frags, 4 v_wmma per K-step.
//   A-frag (16x32): lane m = l&15; K chunks at 8*hi and 16+8*hi.
//   B-frag (32x16): lane n = l&15; K chunks at 8*hi and 16+8*hi.
//   D-frag (16x16): lane n = l&15; rows v+8*hi for VGPR v.
// ---------------------------------------------------------------------------
__global__ __launch_bounds__(256) void gemm_bf16_kernel(
    const __bf16* __restrict__ A, const __bf16* __restrict__ Bm,
    float* __restrict__ C, const float* __restrict__ bias,
    int Mdim, int Ndim, int Kdim) {
  __shared__ __align__(16) __bf16 Abuf[2][64][32];
  __shared__ __align__(16) __bf16 Bbuf[2][128][32];

  const int tid  = threadIdx.x;
  const int lane = tid & 31;
  const int wave = tid >> 5;
  const int wr = wave >> 2;        // 0..1  (32-row strip)
  const int wc = wave & 3;         // 0..3  (32-col strip)
  const int nlo = lane & 15, hi = lane >> 4;

  const int ntiles = Ndim >> 7;    // macro-cols of 128
  const int tn = blockIdx.x % ntiles;
  const int tm = blockIdx.x / ntiles;

  // staging assignment: thread -> one A row-chunk + two B row-chunks (16B each)
  const int lr = tid >> 2;         // 0..63
  const int lc = (tid & 3) * 8;    // k element offset within 32
  const __bf16* agsrc  = A  + (size_t)(tm * 64 + lr) * Kdim + lc;
  const __bf16* bgsrc0 = Bm + (size_t)(tn * 128 + lr) * Kdim + lc;
  const __bf16* bgsrc1 = Bm + (size_t)(tn * 128 + 64 + lr) * Kdim + lc;
  const unsigned lA  = (unsigned)(uintptr_t)&Abuf[0][lr][lc];
  const unsigned lB0 = (unsigned)(uintptr_t)&Bbuf[0][lr][lc];
  const unsigned lB1 = (unsigned)(uintptr_t)&Bbuf[0][64 + lr][lc];
  const unsigned bufStrideA = (unsigned)(64 * 32 * 2);
  const unsigned bufStrideB = (unsigned)(128 * 32 * 2);

  const int nsteps = Kdim >> 5;

  // prologue: stage k-step 0 into buffer 0
  async_ld_b128(lA, (unsigned long long)(uintptr_t)agsrc);
  async_ld_b128(lB0, (unsigned long long)(uintptr_t)bgsrc0);
  async_ld_b128(lB1, (unsigned long long)(uintptr_t)bgsrc1);
  wait_async0();
  __syncthreads();

  union F { v16bf v; v8bf h[2]; };
  v8f acc00 = {}, acc01 = {}, acc10 = {}, acc11 = {};

  for (int s = 0; s < nsteps; ++s) {
    const int buf = s & 1;
    if (s + 1 < nsteps) {  // prefetch next K-step into other buffer
      const int k1 = (s + 1) << 5;
      const unsigned nb = (unsigned)(buf ^ 1);
      async_ld_b128(lA + nb * bufStrideA,
                    (unsigned long long)(uintptr_t)(agsrc + k1));
      async_ld_b128(lB0 + nb * bufStrideB,
                    (unsigned long long)(uintptr_t)(bgsrc0 + k1));
      async_ld_b128(lB1 + nb * bufStrideB,
                    (unsigned long long)(uintptr_t)(bgsrc1 + k1));
    }

    // fragments from LDS
    F a0, a1, b0, b1;
    const __bf16* ar0 = &Abuf[buf][wr * 32 + nlo][0];
    const __bf16* ar1 = &Abuf[buf][wr * 32 + 16 + nlo][0];
    a0.h[0] = *(const v8bf*)(ar0 + 8 * hi);
    a0.h[1] = *(const v8bf*)(ar0 + 16 + 8 * hi);
    a1.h[0] = *(const v8bf*)(ar1 + 8 * hi);
    a1.h[1] = *(const v8bf*)(ar1 + 16 + 8 * hi);
    const __bf16* br0 = &Bbuf[buf][wc * 32 + nlo][0];
    const __bf16* br1 = &Bbuf[buf][wc * 32 + 16 + nlo][0];
    b0.h[0] = *(const v8bf*)(br0 + 8 * hi);
    b0.h[1] = *(const v8bf*)(br0 + 16 + 8 * hi);
    b1.h[0] = *(const v8bf*)(br1 + 8 * hi);
    b1.h[1] = *(const v8bf*)(br1 + 16 + 8 * hi);

    acc00 = __builtin_amdgcn_wmma_f32_16x16x32_bf16(false, a0.v, false, b0.v,
                                                    (short)0, acc00, false, false);
    acc01 = __builtin_amdgcn_wmma_f32_16x16x32_bf16(false, a0.v, false, b1.v,
                                                    (short)0, acc01, false, false);
    acc10 = __builtin_amdgcn_wmma_f32_16x16x32_bf16(false, a1.v, false, b0.v,
                                                    (short)0, acc10, false, false);
    acc11 = __builtin_amdgcn_wmma_f32_16x16x32_bf16(false, a1.v, false, b1.v,
                                                    (short)0, acc11, false, false);

    wait_async0();     // own next-step DMAs complete
    __syncthreads();   // everyone done reading current buffer + all DMAs visible
  }

  const int r0 = tm * 64 + wr * 32;
  const int n0 = tn * 128 + wc * 32;
  const float bia0 = bias ? bias[n0 + nlo] : 0.f;
  const float bia1 = bias ? bias[n0 + 16 + nlo] : 0.f;
#pragma unroll
  for (int v = 0; v < 8; ++v) {
    const int rA = r0 + v + 8 * hi;
    const int rB = rA + 16;
    C[(size_t)rA * Ndim + n0 + nlo]      = acc00[v] + bia0;
    C[(size_t)rA * Ndim + n0 + 16 + nlo] = acc01[v] + bia1;
    C[(size_t)rB * Ndim + n0 + nlo]      = acc10[v] + bia0;
    C[(size_t)rB * Ndim + n0 + 16 + nlo] = acc11[v] + bia1;
  }
}

// ---------------------------------------------------------------------------
// Split qkv (f32 [B*N, 3D]) into bf16 q[b,h,n,d], k[b,h,n,d], vT[b,h,d,n]
// ---------------------------------------------------------------------------
__global__ void split_qkv_kernel(const float* __restrict__ qkv,
                                 __bf16* __restrict__ qbf,
                                 __bf16* __restrict__ kbf,
                                 __bf16* __restrict__ vtbf) {
  int i = blockIdx.x * blockDim.x + threadIdx.x;  // B*H*N*HD
  int d = i & (HD_ - 1);
  int t = i >> 6;
  int n = t % N_; t /= N_;
  int h = t % H_;
  int b = t / H_;
  size_t row = ((size_t)(b * N_ + n)) * (3 * D_) + h * HD_ + d;
  float qv = qkv[row];
  float kv = qkv[row + D_];
  float vv = qkv[row + 2 * D_];
  size_t qi = (((size_t)(b * H_ + h)) * N_ + n) * HD_ + d;
  qbf[qi] = (__bf16)qv;
  kbf[qi] = (__bf16)kv;
  size_t vi = (((size_t)(b * H_ + h)) * HD_ + d) * N_ + n;
  vtbf[vi] = (__bf16)vv;
}

// ---------------------------------------------------------------------------
// Coarse block means qb, kb [B,H,M,HD] (f32) from qkv f32 buffer
// ---------------------------------------------------------------------------
__global__ void block_mean_kernel(const float* __restrict__ qkv,
                                  float* __restrict__ qb,
                                  float* __restrict__ kb) {
  int i = blockIdx.x * blockDim.x + threadIdx.x;  // B*H*M*HD
  int d = i & (HD_ - 1);
  int t = i >> 6;
  int m = t % M_; t /= M_;
  int h = t % H_;
  int b = t / H_;
  float sq = 0.f, sk = 0.f;
  for (int tt = 0; tt < BS_; ++tt) {
    size_t row = ((size_t)(b * N_ + m * BS_ + tt)) * (3 * D_) + h * HD_ + d;
    sq += qkv[row];
    sk += qkv[row + D_];
  }
  size_t o = (((size_t)(b * H_ + h)) * M_ + m) * HD_ + d;
  qb[o] = sq * (1.f / BS_);
  kb[o] = sk * (1.f / BS_);
}

// ---------------------------------------------------------------------------
// Block scores + top-2 threshold + diagonal -> 32-bit keep mask per (b,h,m).
// One wave32 per row; lane = key block. M_==32 fits exactly in one u32 ballot.
// ---------------------------------------------------------------------------
__global__ __launch_bounds__(128) void block_mask_kernel(
    const float* __restrict__ qb, const float* __restrict__ kb,
    unsigned* __restrict__ mask) {
  int lane = threadIdx.x & 31;
  int wflat = (blockIdx.x * blockDim.x + threadIdx.x) >> 5;  // (b*H+h)*M + m
  int m = wflat % M_;
  int bh = wflat / M_;
  const float* qrow = qb + ((size_t)bh * M_ + m) * HD_;
  const float* krow = kb + ((size_t)bh * M_ + lane) * HD_;
  float s = 0.f;
  for (int d = 0; d < HD_; ++d) s += qrow[d] * krow[d];
  s *= SCALE_;
  // wave max
  float m1 = s;
  for (int off = 16; off; off >>= 1) m1 = fmaxf(m1, __shfl_xor(m1, off));
  unsigned long long bal = __ballot(s == m1);
  int firstl = __ffsll(bal) - 1;
  // second-largest = threshold for k_keep == 2
  float s2 = (lane == firstl) ? -3.0e38f : s;
  float m2 = s2;
  for (int off = 16; off; off >>= 1) m2 = fmaxf(m2, __shfl_xor(m2, off));
  unsigned keep = (unsigned)__ballot((s >= m2) || (lane == m));
  if (lane == 0) mask[wflat] = keep;
}

// ---------------------------------------------------------------------------
// Sparse flash attention over kept key blocks. One workgroup = one
// (b,h,qblock); 4 waves, each owning a 16-query strip. All matmuls via WMMA;
// online softmax with stats per row; output written bf16 in [B*N, D] layout.
// ---------------------------------------------------------------------------
__global__ __launch_bounds__(128) void sparse_attn_kernel(
    const __bf16* __restrict__ qbf, const __bf16* __restrict__ kbf,
    const __bf16* __restrict__ vtbf, const unsigned* __restrict__ mask,
    __bf16* __restrict__ attnbf) {
  __shared__ __align__(16) float  Sbuf[4][16][64];
  __shared__ __align__(16) __bf16 Pbuf[4][16][64];
  __shared__ float alphaBuf[4][16];
  __shared__ float lBuf[4][16];

  const int lane = threadIdx.x & 31;
  const int wave = threadIdx.x >> 5;
  const int nlo = lane & 15, hi = lane >> 4;
  const int bhm = blockIdx.x;
  const int m = bhm % M_;
  const int bh = bhm / M_;
  const unsigned mk = mask[bhm];

  union F { v16bf v; v8bf h[2]; };

  // Q fragments (row-strip fixed per wave, reused across all key blocks)
  const __bf16* qrow =
      qbf + (((size_t)bh) * N_ + m * BS_ + wave * 16 + nlo) * HD_;
  F aq0, aq1;
  aq0.h[0] = *(const v8bf*)(qrow + 8 * hi);
  aq0.h[1] = *(const v8bf*)(qrow + 16 + 8 * hi);
  aq1.h[0] = *(const v8bf*)(qrow + 32 + 8 * hi);
  aq1.h[1] = *(const v8bf*)(qrow + 48 + 8 * hi);

  float mi = -3.0e38f, li = 0.f;
  v8f o0 = {}, o1 = {}, o2 = {}, o3 = {};

  unsigned mm = mk;
  while (mm) {
    const int j = __ffs(mm) - 1;
    mm &= mm - 1;
    const int n0 = j * BS_;

    // ---- S = Q K^T * scale (16x64 strip, 4 col-tiles x 2 k-steps) ----
#pragma unroll
    for (int c = 0; c < 4; ++c) {
      const __bf16* krow =
          kbf + (((size_t)bh) * N_ + n0 + c * 16 + nlo) * HD_;
      F bk0, bk1;
      bk0.h[0] = *(const v8bf*)(krow + 8 * hi);
      bk0.h[1] = *(const v8bf*)(krow + 16 + 8 * hi);
      bk1.h[0] = *(const v8bf*)(krow + 32 + 8 * hi);
      bk1.h[1] = *(const v8bf*)(krow + 48 + 8 * hi);
      v8f s = {};
      s = __builtin_amdgcn_wmma_f32_16x16x32_bf16(false, aq0.v, false, bk0.v,
                                                  (short)0, s, false, false);
      s = __builtin_amdgcn_wmma_f32_16x16x32_bf16(false, aq1.v, false, bk1.v,
                                                  (short)0, s, false, false);
#pragma unroll
      for (int v = 0; v < 8; ++v)
        Sbuf[wave][v + 8 * hi][c * 16 + nlo] = s[v] * SCALE_;
    }
    __syncthreads();

    // ---- online softmax: lane pair (nlo, nlo+16) owns row nlo ----
    {
      const int r = nlo;
      const int c0 = 32 * hi;
      float lmax = -3.0e38f;
      for (int c = 0; c < 32; ++c) lmax = fmaxf(lmax, Sbuf[wave][r][c0 + c]);
      lmax = fmaxf(lmax, __shfl_xor(lmax, 16));
      const float mnew = fmaxf(mi, lmax);
      const float alpha = __expf(mi - mnew);
      float lsum = 0.f;
      for (int c = 0; c < 32; ++c) {
        float p = __expf(Sbuf[wave][r][c0 + c] - mnew);
        Pbuf[wave][r][c0 + c] = (__bf16)p;
        lsum += p;
      }
      lsum += __shfl_xor(lsum, 16);
      li = li * alpha + lsum;
      mi = mnew;
      if (hi == 0) alphaBuf[wave][r] = alpha;
    }
    __syncthreads();

    // ---- rescale O by per-row alpha ----
    float a8[8];
#pragma unroll
    for (int v = 0; v < 8; ++v) a8[v] = alphaBuf[wave][v + 8 * hi];
#pragma unroll
    for (int v = 0; v < 8; ++v) {
      o0[v] *= a8[v]; o1[v] *= a8[v]; o2[v] *= a8[v]; o3[v] *= a8[v];
    }

    // ---- O += P V  (P from LDS as A-frags; V^T from global as B-frags) ----
    F ap0, ap1;
    ap0.h[0] = *(const v8bf*)&Pbuf[wave][nlo][8 * hi];
    ap0.h[1] = *(const v8bf*)&Pbuf[wave][nlo][16 + 8 * hi];
    ap1.h[0] = *(const v8bf*)&Pbuf[wave][nlo][32 + 8 * hi];
    ap1.h[1] = *(const v8bf*)&Pbuf[wave][nlo][48 + 8 * hi];
#pragma unroll
    for (int c = 0; c < 4; ++c) {
      const __bf16* vrow =
          vtbf + (((size_t)bh) * HD_ + c * 16 + nlo) * N_ + n0;
      F bv0, bv1;
      bv0.h[0] = *(const v8bf*)(vrow + 8 * hi);
      bv0.h[1] = *(const v8bf*)(vrow + 16 + 8 * hi);
      bv1.h[0] = *(const v8bf*)(vrow + 32 + 8 * hi);
      bv1.h[1] = *(const v8bf*)(vrow + 48 + 8 * hi);
      v8f* oc = (c == 0) ? &o0 : (c == 1) ? &o1 : (c == 2) ? &o2 : &o3;
      *oc = __builtin_amdgcn_wmma_f32_16x16x32_bf16(false, ap0.v, false, bv0.v,
                                                    (short)0, *oc, false, false);
      *oc = __builtin_amdgcn_wmma_f32_16x16x32_bf16(false, ap1.v, false, bv1.v,
                                                    (short)0, *oc, false, false);
    }
    __syncthreads();
  }

  if (hi == 0) lBuf[wave][nlo] = li;
  __syncthreads();

  const int b = bh / H_, h = bh % H_;
#pragma unroll
  for (int v = 0; v < 8; ++v) {
    const int q = v + 8 * hi;
    const float inv = 1.f / lBuf[wave][q];
    const int nq = m * BS_ + wave * 16 + q;
    const size_t base = ((size_t)(b * N_ + nq)) * D_ + h * HD_;
    attnbf[base + 0 + nlo]  = (__bf16)(o0[v] * inv);
    attnbf[base + 16 + nlo] = (__bf16)(o1[v] * inv);
    attnbf[base + 32 + nlo] = (__bf16)(o2[v] * inv);
    attnbf[base + 48 + nlo] = (__bf16)(o3[v] * inv);
  }
}

// ---------------------------------------------------------------------------
// Host-side orchestration
// ---------------------------------------------------------------------------
extern "C" void kernel_launch(void* const* d_in, const int* in_sizes, int n_in,
                              void* d_out, int out_size, void* d_ws,
                              size_t ws_size, hipStream_t stream) {
  (void)in_sizes; (void)n_in; (void)out_size; (void)ws_size;
  const float* x      = (const float*)d_in[0];  // [B,N,D]
  const float* qkv_w  = (const float*)d_in[1];  // [3D,D]
  const float* proj_w = (const float*)d_in[2];  // [D,D]
  const float* proj_b = (const float*)d_in[3];  // [D]
  float* out = (float*)d_out;                   // [B,N,D]

  char* ws = (char*)d_ws;
  size_t off = 0;
  auto alloc = [&](size_t bytes) -> void* {
    off = (off + 255) & ~(size_t)255;
    void* p = ws + off;
    off += bytes;
    return p;
  };

  const int rows = B_ * N_;                           // 4096
  __bf16* xbf     = (__bf16*)alloc((size_t)rows * D_ * 2);
  __bf16* qkvwbf  = (__bf16*)alloc((size_t)3 * D_ * D_ * 2);
  __bf16* projwbf = (__bf16*)alloc((size_t)D_ * D_ * 2);
  float*  qkvf    = (float*)alloc((size_t)rows * 3 * D_ * 4);
  __bf16* qbf     = (__bf16*)alloc((size_t)B_ * H_ * N_ * HD_ * 2);
  __bf16* kbf     = (__bf16*)alloc((size_t)B_ * H_ * N_ * HD_ * 2);
  __bf16* vtbf    = (__bf16*)alloc((size_t)B_ * H_ * HD_ * N_ * 2);
  float*  qb      = (float*)alloc((size_t)B_ * H_ * M_ * HD_ * 4);
  float*  kb      = (float*)alloc((size_t)B_ * H_ * M_ * HD_ * 4);
  unsigned* bmask = (unsigned*)alloc((size_t)B_ * H_ * M_ * 4);
  __bf16* attnbf  = (__bf16*)alloc((size_t)rows * D_ * 2);

  // 1) convert inputs to bf16
  cvt_bf16_kernel<<<(rows * D_) / 256, 256, 0, stream>>>(x, xbf, rows * D_);
  cvt_bf16_kernel<<<(3 * D_ * D_) / 256, 256, 0, stream>>>(qkv_w, qkvwbf,
                                                           3 * D_ * D_);
  cvt_bf16_kernel<<<(D_ * D_) / 256, 256, 0, stream>>>(proj_w, projwbf,
                                                       D_ * D_);
  // 2) qkv = x @ qkv_w^T   (M=4096, N=3072, K=1024); 64x128 macro-tiles
  {
    int blocks = (rows / 64) * ((3 * D_) / 128);
    gemm_bf16_kernel<<<blocks, 256, 0, stream>>>(xbf, qkvwbf, qkvf, nullptr,
                                                 rows, 3 * D_, D_);
  }
  // 3) split into bf16 q,k,vT
  split_qkv_kernel<<<(B_ * H_ * N_ * HD_) / 256, 256, 0, stream>>>(qkvf, qbf,
                                                                   kbf, vtbf);
  // 4) coarse block means
  block_mean_kernel<<<(B_ * H_ * M_ * HD_) / 256, 256, 0, stream>>>(qkvf, qb,
                                                                    kb);
  // 5) top-2 block mask (one wave per (b,h,m))
  block_mask_kernel<<<(B_ * H_ * M_ * 32) / 128, 128, 0, stream>>>(qb, kb,
                                                                   bmask);
  // 6) sparse flash attention
  sparse_attn_kernel<<<B_ * H_ * M_, 128, 0, stream>>>(qbf, kbf, vtbf, bmask,
                                                       attnbf);
  // 7) out = attn @ proj_w^T + proj_b  (M=4096, N=1024, K=1024)
  {
    int blocks = (rows / 64) * (D_ / 128);
    gemm_bf16_kernel<<<blocks, 256, 0, stream>>>(attnbf, projwbf, out,
                                                 proj_b, rows, D_, D_);
  }
}
